// MultiHeadSelfAttention_73864847556611
// MI455X (gfx1250) — matrix operand order repaired and checked by reading, not compiled, not run on hardware
//
#include <hip/hip_runtime.h>
#include <hip/hip_bf16.h>
#include <stdint.h>

typedef __attribute__((ext_vector_type(16))) _Float16     v16h;
typedef __attribute__((ext_vector_type(8)))  _Float16     v8h;
typedef __attribute__((ext_vector_type(4)))  _Float16     v4h;
typedef __attribute__((ext_vector_type(8)))  float        v8f;
typedef __attribute__((ext_vector_type(4)))  float        vf4;
typedef __attribute__((ext_vector_type(8)))  unsigned int v8u;
typedef __attribute__((ext_vector_type(4)))  unsigned int v4u;

#define D_MODEL   1024
#define NUM_HEADS 16
#define HEAD_DIM  64
#define BATCH     4
#define SEQ       2048
#define ROWS      (BATCH * SEQ)   // 8192

static __device__ __forceinline__ v8f wmma_f16(const v8u& a, const v8u& b, v8f c) {
  return __builtin_amdgcn_wmma_f32_16x16x32_f16(
      false, __builtin_bit_cast(v16h, a),
      false, __builtin_bit_cast(v16h, b),
      (short)0, c, false, false);
}

// ds_swizzle_b32 XOR-butterfly (group-of-32 encoding: xor[14:10], and=0x1F)
template<int IMM>
static __device__ __forceinline__ float ds_swz(float v) {
  return __builtin_bit_cast(float,
      __builtin_amdgcn_ds_swizzle(__builtin_bit_cast(int, v), IMM));
}
#define SWZ_X1 0x041F
#define SWZ_X2 0x081F
#define SWZ_X4 0x101F
#define SWZ_X8 0x201F

// ---------------------------------------------------------------------------
// Tiled GEMM: Out[M,N] = A[M,K] * W[N,K]^T   (x @ W.T convention)
// Block tile 128x128, BK=32; 256 threads = 8 waves (2 down x 4 across);
// each wave computes a 64x32 C tile = 8 WMMA per K-step (4 A-frags x 2 B-frags).
// ---------------------------------------------------------------------------
template<bool AHALF, bool OUTF>
__global__ __launch_bounds__(256)
void gemm_xwT(const float* __restrict__ A32, const _Float16* __restrict__ A16,
              const float* __restrict__ W,
              _Float16* __restrict__ outH, float* __restrict__ outF)
{
  __shared__ _Float16 sA[128 * 32];   // [row][k]   8 KB
  __shared__ _Float16 sBt[32 * 128];  // [k][n]     8 KB (W transposed)

  const int tid  = threadIdx.x;
  const int lane = tid & 31;
  const int w    = tid >> 5;
  const int wr   = (w & 1) * 64;   // wave row offset within 128-row block
  const int wc   = (w >> 1) * 32;  // wave col offset within 128-col block
  const int m    = lane & 15;
  const int hi   = lane >> 4;
  const int rowBase = blockIdx.x * 128;
  const int colBase = blockIdx.y * 128;

  v8f acc[4][2];
  v8f zero = {};
  #pragma unroll
  for (int t = 0; t < 4; ++t)
    #pragma unroll
    for (int j = 0; j < 2; ++j) acc[t][j] = zero;

  for (int kt = 0; kt < D_MODEL; kt += 32) {
    if (AHALF) {
      #pragma unroll
      for (int i = 0; i < 2; ++i) {       // 128 rows x 4 chunks of 8 f16
        int idx = tid + i * 256;
        int r = idx >> 2, c = idx & 3;
        v4u d = *(const v4u*)(A16 + (size_t)(rowBase + r) * D_MODEL + kt + c * 8);
        *(v4u*)(sA + r * 32 + c * 8) = d;
      }
    } else {
      #pragma unroll
      for (int i = 0; i < 4; ++i) {       // 128 rows x 8 chunks of 4 f32
        int idx = tid + i * 256;
        int r = idx >> 3, c = idx & 7;
        vf4 fv = *(const vf4*)(A32 + (size_t)(rowBase + r) * D_MODEL + kt + c * 4);
        v4h hv;
        hv[0] = (_Float16)fv[0]; hv[1] = (_Float16)fv[1];
        hv[2] = (_Float16)fv[2]; hv[3] = (_Float16)fv[3];
        *(v4h*)(sA + r * 32 + c * 4) = hv;
      }
    }
    #pragma unroll
    for (int i = 0; i < 4; ++i) {         // W tile 128(N)x32(K) -> transposed
      int idx = tid + i * 256;
      int e = idx >> 3, d4 = (idx & 7) * 4;
      vf4 wv = *(const vf4*)(W + (size_t)(colBase + e) * D_MODEL + kt + d4);
      #pragma unroll
      for (int k2 = 0; k2 < 4; ++k2)
        sBt[(d4 + k2) * 128 + e] = (_Float16)wv[k2];
    }
    __syncthreads();

    v8u a[4], b[2];
    #pragma unroll
    for (int t = 0; t < 4; ++t) {         // A-frag = two b128 LDS loads
      const _Float16* arow = sA + (wr + t * 16 + m) * 32;
      v4u lo = *(const v4u*)(arow + hi * 8);
      v4u h4 = *(const v4u*)(arow + 16 + hi * 8);
      #pragma unroll
      for (int r = 0; r < 4; ++r) { a[t][r] = lo[r]; a[t][r + 4] = h4[r]; }
    }
    #pragma unroll
    for (int j = 0; j < 2; ++j)
      #pragma unroll
      for (int r = 0; r < 8; ++r)
        b[j][r] = (unsigned int)*(const unsigned short*)(sBt + (hi * 8 + 2 * r + (r >> 2) * 8) * 128 + wc + j * 16 + m) | ((unsigned int)*(const unsigned short*)(sBt + (hi * 8 + 2 * r + (r >> 2) * 8 + 1) * 128 + wc + j * 16 + m) << 16);

    #pragma unroll
    for (int t = 0; t < 4; ++t)
      #pragma unroll
      for (int j = 0; j < 2; ++j)
        acc[t][j] = wmma_f16(a[t], b[j], acc[t][j]);
    __syncthreads();
  }

  #pragma unroll
  for (int t = 0; t < 4; ++t)
    #pragma unroll
    for (int j = 0; j < 2; ++j)
      #pragma unroll
      for (int g = 0; g < 8; ++g) {
        size_t row = (size_t)(rowBase + wr + t * 16 + g + 8 * hi);
        int    col = colBase + wc + j * 16 + (lane & 15);
        if (OUTF) outF[row * D_MODEL + col] = acc[t][j][g];
        else      outH[row * D_MODEL + col] = (_Float16)acc[t][j][g];
      }
}

// ---------------------------------------------------------------------------
// Flash attention: per (batch, head, 64-query tile). 128 threads = 4 waves,
// each wave owns 16 query rows. Keys processed in causal 32-wide tiles.
// V tile staged with CDNA5 async global->LDS DMA (ASYNCcnt).
// ---------------------------------------------------------------------------
__global__ __launch_bounds__(128)
void flash_attn(const _Float16* __restrict__ Qh, const _Float16* __restrict__ Kh,
                const _Float16* __restrict__ Vh, _Float16* __restrict__ Ctx)
{
  __shared__ _Float16 sKt[64 * 32];     // [dh][key]  (K transposed)
  __shared__ _Float16 sV [32 * 64];     // [key][dh]
  __shared__ _Float16 sP [4][16 * 32];  // per-wave P scratch (C->A relayout)

  const int tid  = threadIdx.x;
  const int lane = tid & 31;
  const int w    = tid >> 5;
  const int m    = lane & 15;
  const int hi   = lane >> 4;
  const int b    = blockIdx.z;
  const int h    = blockIdx.y;
  const int q0   = blockIdx.x * 64;
  const int qw0  = q0 + w * 16;
  const size_t rowOff = (size_t)b * SEQ;
  const int headOff   = h * HEAD_DIM;
  const float scale   = 0.125f;          // 1/sqrt(64)

  // Q fragments (16 rows x 64 dh -> two 16x32 A-fragments), 2x b128 each
  v8u qf[2];
  #pragma unroll
  for (int c = 0; c < 2; ++c) {
    const _Float16* qrow = Qh + (rowOff + qw0 + m) * D_MODEL + headOff + c * 32;
    v4u lo = *(const v4u*)(qrow + hi * 8);
    v4u h4 = *(const v4u*)(qrow + 16 + hi * 8);
    #pragma unroll
    for (int r = 0; r < 4; ++r) { qf[c][r] = lo[r]; qf[c][r + 4] = h4[r]; }
  }

  float mrow[8], lrow[8];
  v8f acc[4];
  v8f zero = {};
  #pragma unroll
  for (int g = 0; g < 8; ++g) { mrow[g] = -1.0e30f; lrow[g] = 0.0f; }
  #pragma unroll
  for (int t = 0; t < 4; ++t) acc[t] = zero;

  for (int kb = 0; kb < q0 + 64; kb += 32) {
    // prefetch next K/V tiles into cache hierarchy
    if (kb + 32 < q0 + 64) {
      const _Float16* pk = Kh + (rowOff + kb + 32 + lane) * D_MODEL + headOff;
      const _Float16* pv = Vh + (rowOff + kb + 32 + lane) * D_MODEL + headOff;
      __builtin_prefetch(pk, 0, 1);
      __builtin_prefetch(pv, 0, 1);
    }
    // --- V tile via async DMA: 32 rows x 128B = 256 x 16B chunks, 2/thread ---
    #pragma unroll
    for (int i = 0; i < 2; ++i) {
      int idx = tid + i * 128;            // 0..255
      int kr = idx >> 3, c = idx & 7;     // row, 16B chunk in row
      unsigned ldsOff = (unsigned)(uintptr_t)(&sV[kr * 64 + c * 8]);
      unsigned long long ga = (unsigned long long)(uintptr_t)
          (Vh + (rowOff + kb + kr) * D_MODEL + headOff + c * 8);
      asm volatile("global_load_async_to_lds_b128 %0, %1, off"
                   :: "v"(ldsOff), "v"(ga) : "memory");
    }
    // --- K tile: b128 global loads, transposed scatter into sKt[dh][key] ---
    #pragma unroll
    for (int i = 0; i < 2; ++i) {
      int idx = tid + i * 128;            // 0..255 : 32 rows x 8 chunks of 8 f16
      int kr = idx >> 3, c = idx & 7;
      v8h kv = *(const v8h*)(Kh + (rowOff + kb + kr) * D_MODEL + headOff + c * 8);
      #pragma unroll
      for (int e = 0; e < 8; ++e)
        sKt[(c * 8 + e) * 32 + kr] = kv[e];
    }
    asm volatile("s_wait_asynccnt 0" ::: "memory");
    __syncthreads();

    if (kb <= qw0 + 15) {  // wave-level causal skip
      // ---- scores: two 16x16 tiles over keys kb..kb+31 ----
      v8f sc[2];
      #pragma unroll
      for (int t = 0; t < 2; ++t) {
        v8f s = zero;
        #pragma unroll
        for (int c = 0; c < 2; ++c) {
          v8u bk;  // B 32x16: lane%16 = key, VGPR r -> dh pair (table order)
          #pragma unroll
          for (int r = 0; r < 8; ++r)
            bk[r] = (unsigned int)*(const unsigned short*)(sKt + (c * 32 + hi * 8 + 2 * r + (r >> 2) * 8) * 32 + t * 16 + m) | ((unsigned int)*(const unsigned short*)(sKt + (c * 32 + hi * 8 + 2 * r + (r >> 2) * 8 + 1) * 32 + t * 16 + m) << 16);
          s = wmma_f16(qf[c], bk, s);
        }
        sc[t] = s;
      }
      const bool needMask = (kb + 31 > qw0);   // wave-uniform: diagonal tile?
      // ---- mask + online softmax (row = g + 8*hi, col = lane&15) ----
      #pragma unroll
      for (int g = 0; g < 8; ++g) {
        float s0 = sc[0][g] * scale;
        float s1 = sc[1][g] * scale;
        if (needMask) {
          int q   = qw0 + g + 8 * hi;
          int k0i = kb + (lane & 15);
          s0 = (k0i      <= q) ? s0 : -1.0e30f;
          s1 = (k0i + 16 <= q) ? s1 : -1.0e30f;
        }
        float mx = fmaxf(s0, s1);
        mx = fmaxf(mx, ds_swz<SWZ_X1>(mx));
        mx = fmaxf(mx, ds_swz<SWZ_X2>(mx));
        mx = fmaxf(mx, ds_swz<SWZ_X4>(mx));
        mx = fmaxf(mx, ds_swz<SWZ_X8>(mx));
        float mnew = fmaxf(mrow[g], mx);
        float corr = __expf(mrow[g] - mnew);
        mrow[g] = mnew;
        float p0 = __expf(s0 - mnew);
        float p1 = __expf(s1 - mnew);
        float rs = p0 + p1;
        rs += ds_swz<SWZ_X1>(rs);
        rs += ds_swz<SWZ_X2>(rs);
        rs += ds_swz<SWZ_X4>(rs);
        rs += ds_swz<SWZ_X8>(rs);
        lrow[g] = lrow[g] * corr + rs;
        #pragma unroll
        for (int t = 0; t < 4; ++t) acc[t][g] *= corr;
        int prow = g + 8 * hi;                       // P to row-major 16x32
        sP[w][prow * 32 +      (lane & 15)] = (_Float16)p0;
        sP[w][prow * 32 + 16 + (lane & 15)] = (_Float16)p1;
      }
      asm volatile("s_wait_dscnt 0" ::: "memory");   // wave-local LDS relayout
      v8u pf;
      {
        const _Float16* prow = &sP[w][m * 32];
        v4u lo = *(const v4u*)(prow + hi * 8);
        v4u h4 = *(const v4u*)(prow + 16 + hi * 8);
        #pragma unroll
        for (int r = 0; r < 4; ++r) { pf[r] = lo[r]; pf[r + 4] = h4[r]; }
      }
      // ---- acc += P(16x32) * V(32x64) : four 16x16 output tiles ----
      #pragma unroll
      for (int t = 0; t < 4; ++t) {
        v8u bv;  // B 32x16: lane%16 = dh, VGPR r -> key pair (table order)
        #pragma unroll
        for (int r = 0; r < 8; ++r)
          bv[r] = (unsigned int)*(const unsigned short*)(sV + (hi * 8 + 2 * r + (r >> 2) * 8) * 64 + t * 16 + m) | ((unsigned int)*(const unsigned short*)(sV + (hi * 8 + 2 * r + (r >> 2) * 8 + 1) * 64 + t * 16 + m) << 16);
        acc[t] = wmma_f16(pf, bv, acc[t]);
      }
    }
    __syncthreads();
  }

  #pragma unroll
  for (int g = 0; g < 8; ++g) {
    float inv = 1.0f / lrow[g];
    size_t row = rowOff + qw0 + g + 8 * hi;
    #pragma unroll
    for (int t = 0; t < 4; ++t) {
      int col = headOff + t * 16 + (lane & 15);
      Ctx[row * D_MODEL + col] = (_Float16)(acc[t][g] * inv);
    }
  }
}

// ---------------------------------------------------------------------------
extern "C" void kernel_launch(void* const* d_in, const int* in_sizes, int n_in,
                              void* d_out, int out_size, void* d_ws, size_t ws_size,
                              hipStream_t stream)
{
  const float* x  = (const float*)d_in[0];
  const float* Wq = (const float*)d_in[1];
  const float* Wk = (const float*)d_in[2];
  const float* Wv = (const float*)d_in[3];
  const float* Wo = (const float*)d_in[4];
  float* out = (float*)d_out;

  const size_t NM = (size_t)ROWS * D_MODEL;  // 8192*1024 f16 elems = 16 MB each
  _Float16* Qh = (_Float16*)d_ws;
  _Float16* Kh = Qh + NM;
  _Float16* Vh = Kh + NM;
  _Float16* Ch = Vh + NM;                    // total 64 MB workspace

  dim3 gG(ROWS / 128, D_MODEL / 128, 1);
  gemm_xwT<false, false><<<gG, 256, 0, stream>>>(x, nullptr, Wq, Qh, nullptr);
  gemm_xwT<false, false><<<gG, 256, 0, stream>>>(x, nullptr, Wk, Kh, nullptr);
  gemm_xwT<false, false><<<gG, 256, 0, stream>>>(x, nullptr, Wv, Vh, nullptr);

  dim3 gA(SEQ / 64, NUM_HEADS, BATCH);
  flash_attn<<<gA, 128, 0, stream>>>(Qh, Kh, Vh, Ch);

  gemm_xwT<true, true><<<gG, 256, 0, stream>>>(nullptr, Ch, Wo, nullptr, out);
}
